// SAGE_Classifier_43404939493469
// MI455X (gfx1250) — matrix operand order repaired and checked by reading, not compiled, not run on hardware
//
#include <hip/hip_runtime.h>

#define DEV __device__ __forceinline__

constexpr int kN  = 32768;
constexpr int kH  = 128;
constexpr int kD  = 16;
constexpr int kB  = 256;
constexpr int kE2 = kN * kD;
constexpr int kNC = 10;
constexpr int kNH = kN * kH;

typedef __attribute__((ext_vector_type(16))) __bf16        v16bf;
typedef __attribute__((ext_vector_type(8)))  float         v8f;
typedef __attribute__((ext_vector_type(8)))  unsigned short us8;

union Frag {
  v16bf bf;
  us8   h[2];
};

DEV unsigned short f2bf(float f) {
  unsigned int u = __builtin_bit_cast(unsigned int, f);
  u += 0x7fffu + ((u >> 16) & 1u);           // round-to-nearest-even
  return (unsigned short)(u >> 16);
}

DEV v8f wmma_bf16(Frag a, Frag b, v8f c) {
  return __builtin_amdgcn_wmma_f32_16x16x32_bf16(false, a.bf, false, b.bf,
                                                 (short)0, c, false, false);
}

// A fragment (16x32 bf16 tile, M = lane%16):
//   lanes 0-15 : elements 0-7 = K 0-7,  elements 8-15 = K 16-23
//   lanes 16-31: elements 0-7 = K 8-15, elements 8-15 = K 24-31
DEV Frag load_a_frag(const unsigned short* base, int ld, int kb, int lane) {
  const unsigned short* p = base + (size_t)(lane & 15) * ld + kb * 32 + ((lane >> 4) << 3);
  Frag f;
  f.h[0] = *(const us8*)(p);
  f.h[1] = *(const us8*)(p + 16);
  return f;
}

// B fragment (32x16 bf16 tile from row-major W[O,K]): lane holds column n=lane%16,
// elements 0-7 = K (lane/16)*8 .. +7, elements 8-15 = K 16+(lane/16)*8 .. +7 of weight row n.
DEV Frag load_b_frag(const unsigned short* wbase, int ld, int kb, int lane) {
  const unsigned short* p = wbase + (size_t)(lane & 15) * ld + kb * 32 + ((lane >> 4) << 3);
  Frag f;
  f.h[0] = *(const us8*)(p);
  f.h[1] = *(const us8*)(p + 16);
  return f;
}

DEV float sigf(float x) { return 1.0f / (1.0f + __expf(-x)); }

// ---------------------------------------------------------------------------
// elementwise helpers
// ---------------------------------------------------------------------------
__global__ void cvt_bf16_kernel(const float* __restrict__ x,
                                unsigned short* __restrict__ y, int n) {
  int i = blockIdx.x * 256 + threadIdx.x;
  if (i < n) y[i] = f2bf(x[i]);
}

__global__ void zero_kernel(float* __restrict__ x, int n) {
  int i = blockIdx.x * 256 + threadIdx.x;
  if (i < n) x[i] = 0.0f;
}

__global__ void embed_kernel(const int* __restrict__ act_ids,
                             const int* __restrict__ dur_ids,
                             const int* __restrict__ act2_ids,
                             const float* __restrict__ emb_act,
                             const float* __restrict__ emb_dur,
                             const float* __restrict__ emb_act2,
                             float* __restrict__ ha, float* __restrict__ hd,
                             unsigned short* __restrict__ concat_bf) {
  int idx = blockIdx.x * 256 + threadIdx.x;       // over N*H
  if (idx >= kNH) return;
  int n = idx >> 7, c = idx & 127;
  float va = emb_act[act_ids[n] * kH + c];
  float vd = emb_dur[dur_ids[n] * kH + c];
  float v2 = emb_act2[act2_ids[n] * kH + c];
  ha[idx] = va;
  hd[idx] = vd;
  concat_bf[(size_t)n * 256 + c]       = f2bf(v2);
  concat_bf[(size_t)n * 256 + 128 + c] = f2bf(vd);
}

__global__ void scatter_add_kernel(const float* __restrict__ m,
                                   const int* __restrict__ src,
                                   const int* __restrict__ dst,
                                   float* __restrict__ agg) {
  size_t idx = (size_t)blockIdx.x * 256 + threadIdx.x;   // over E2*H
  if (idx >= (size_t)kE2 * kH) return;
  int e = (int)(idx >> 7);
  int c = (int)(idx & 127);
  atomicAdd(&agg[(size_t)dst[e] * kH + c], m[(size_t)src[e] * kH + c]);
}

// out = leaky_relu(LayerNorm(x [+ add])) ; one block (128 thr) per row
__global__ __launch_bounds__(128) void ln_lrelu_kernel(const float* __restrict__ x,
                                                       const float* __restrict__ add,
                                                       const float* __restrict__ g,
                                                       const float* __restrict__ b,
                                                       float* __restrict__ out) {
  __shared__ float red[128];
  int row = blockIdx.x, c = threadIdx.x;
  size_t idx = (size_t)row * kH + c;
  float v = x[idx] + (add ? add[idx] : 0.0f);
  red[c] = v;
  __syncthreads();
  for (int s = 64; s > 0; s >>= 1) {
    if (c < s) red[c] += red[c + s];
    __syncthreads();
  }
  float mean = red[0] * (1.0f / kH);
  __syncthreads();
  float d = v - mean;
  red[c] = d * d;
  __syncthreads();
  for (int s = 64; s > 0; s >>= 1) {
    if (c < s) red[c] += red[c + s];
    __syncthreads();
  }
  float var = red[0] * (1.0f / kH);
  float y = d * rsqrtf(var + 1e-5f) * g[c] + b[c];
  out[idx] = (y >= 0.0f) ? y : 0.01f * y;
}

__global__ void seg_accum_kernel(const float* __restrict__ x,
                                 const int* __restrict__ gids,
                                 float* __restrict__ sums,
                                 float* __restrict__ cnt) {
  int idx = blockIdx.x * 256 + threadIdx.x;    // over N*H
  if (idx >= kNH) return;
  int n = idx >> 7, c = idx & 127;
  int g = gids[n];
  atomicAdd(&sums[(size_t)g * kH + c], x[idx]);
  if (cnt && c == 0) atomicAdd(&cnt[g], 1.0f);
}

__global__ void seg_div_kernel(float* __restrict__ sums, const float* __restrict__ cnt) {
  int idx = blockIdx.x * 256 + threadIdx.x;    // over B*H
  if (idx >= kB * kH) return;
  sums[idx] /= cnt[idx >> 7];
}

// tiny exact-fp32 head GEMM: out[M,O] = relu?(X[M,K] @ W[O,K]^T + b)
__global__ void linear_kernel(const float* __restrict__ X, const float* __restrict__ W,
                              const float* __restrict__ b, float* __restrict__ out,
                              int M, int K, int O, int doRelu) {
  int idx = blockIdx.x * 256 + threadIdx.x;
  if (idx >= M * O) return;
  int r = idx / O, o = idx % O;
  const float* xr = X + (size_t)r * K;
  const float* wr = W + (size_t)o * K;
  float acc = b[o];
  for (int k = 0; k < K; ++k) acc = fmaf(xr[k], wr[k], acc);
  if (doRelu) acc = fmaxf(acc, 0.0f);
  out[idx] = acc;
}

__global__ void concat4_kernel(const float* __restrict__ a, const float* __restrict__ b2,
                               const float* __restrict__ c2, const float* __restrict__ d2,
                               float* __restrict__ hc) {
  int idx = blockIdx.x * 256 + threadIdx.x;    // over B*512
  if (idx >= kB * 512) return;
  int r = idx >> 9, c = idx & 511;
  float v;
  if (c < 128)       v = a[r * 128 + c];
  else if (c < 256)  v = b2[r * 128 + c - 128];
  else if (c < 384)  v = c2[r * 128 + c - 256];
  else               v = d2[r * 128 + c - 384];
  hc[idx] = v;
}

// ---------------------------------------------------------------------------
// generic bf16 WMMA GEMM: out[N,Hout] = A[N,K](bf16) @ W[Hout,K]^T(bf16) + bias
// block = 256 thr (8 waves), 32 rows per block; optional bf16 secondary output
// ---------------------------------------------------------------------------
__global__ __launch_bounds__(256) void gemm_bias_kernel(
    const unsigned short* __restrict__ A, const unsigned short* __restrict__ W,
    const float* __restrict__ bias, float* __restrict__ outf,
    unsigned short* __restrict__ outbf, int K, int Hout) {
  const int tid = threadIdx.x, lane = tid & 31, wave = tid >> 5;
  const int row0 = blockIdx.x * 32;
  const int nkb = K >> 5;
  const unsigned short* Arow0 = A + (size_t)row0 * K;
  const unsigned short* Arow1 = A + (size_t)(row0 + 16) * K;
  for (int ct = wave; ct < (Hout >> 4); ct += 8) {
    v8f a0 = (v8f)0.0f, a1 = (v8f)0.0f;
    const unsigned short* Wb = W + (size_t)(ct << 4) * K;
    for (int kb = 0; kb < nkb; ++kb) {
      Frag f0 = load_a_frag(Arow0, K, kb, lane);
      Frag f1 = load_a_frag(Arow1, K, kb, lane);
      Frag wb = load_b_frag(Wb, K, kb, lane);
      a0 = wmma_bf16(f0, wb, a0);
      a1 = wmma_bf16(f1, wb, a1);
    }
    int col = (ct << 4) + (lane & 15);
    float bv = bias[col];
#pragma unroll
    for (int v = 0; v < 8; ++v) {
      int r = v + ((lane >> 4) << 3);
      size_t i0 = (size_t)(row0 + r) * Hout + col;
      size_t i1 = (size_t)(row0 + 16 + r) * Hout + col;
      float x0 = a0[v] + bv, x1 = a1[v] + bv;
      if (outf)  { outf[i0] = x0;        outf[i1] = x1; }
      if (outbf) { outbf[i0] = f2bf(x0); outbf[i1] = f2bf(x1); }
    }
  }
}

// ---------------------------------------------------------------------------
// fused GRU step: gi = a@Wih^T, gh = h@Whh^T, gate math, write f32 + bf16
// ---------------------------------------------------------------------------
__global__ __launch_bounds__(256) void gru_kernel(
    const unsigned short* __restrict__ Abf, const unsigned short* __restrict__ Hbf,
    const float* __restrict__ Hf,
    const unsigned short* __restrict__ Wih, const unsigned short* __restrict__ Whh,
    const float* __restrict__ bih, const float* __restrict__ bhh,
    float* __restrict__ outf, unsigned short* __restrict__ outbf) {
  const int tid = threadIdx.x, lane = tid & 31, wave = tid >> 5;
  const int row0 = blockIdx.x * 32;
  v8f gi[2][3], gh[2][3];
#pragma unroll
  for (int s = 0; s < 2; ++s)
#pragma unroll
    for (int g = 0; g < 3; ++g) { gi[s][g] = (v8f)0.0f; gh[s][g] = (v8f)0.0f; }

  const unsigned short* Ar0 = Abf + (size_t)row0 * kH;
  const unsigned short* Ar1 = Abf + (size_t)(row0 + 16) * kH;
  const unsigned short* Hr0 = Hbf + (size_t)row0 * kH;
  const unsigned short* Hr1 = Hbf + (size_t)(row0 + 16) * kH;
#pragma unroll
  for (int kb = 0; kb < 4; ++kb) {
    Frag aa0 = load_a_frag(Ar0, kH, kb, lane);
    Frag aa1 = load_a_frag(Ar1, kH, kb, lane);
    Frag ah0 = load_a_frag(Hr0, kH, kb, lane);
    Frag ah1 = load_a_frag(Hr1, kH, kb, lane);
#pragma unroll
    for (int g = 0; g < 3; ++g) {
      Frag bi_ = load_b_frag(Wih + (size_t)(g * kH + (wave << 4)) * kH, kH, kb, lane);
      Frag bh_ = load_b_frag(Whh + (size_t)(g * kH + (wave << 4)) * kH, kH, kb, lane);
      gi[0][g] = wmma_bf16(aa0, bi_, gi[0][g]);
      gi[1][g] = wmma_bf16(aa1, bi_, gi[1][g]);
      gh[0][g] = wmma_bf16(ah0, bh_, gh[0][g]);
      gh[1][g] = wmma_bf16(ah1, bh_, gh[1][g]);
    }
  }
  int col = (wave << 4) + (lane & 15);
  float bir = bih[col], biz = bih[kH + col], bin = bih[2 * kH + col];
  float bhr = bhh[col], bhz = bhh[kH + col], bhn = bhh[2 * kH + col];
#pragma unroll
  for (int s = 0; s < 2; ++s)
#pragma unroll
    for (int v = 0; v < 8; ++v) {
      int r = v + ((lane >> 4) << 3) + (s << 4);
      size_t idx = (size_t)(row0 + r) * kH + col;
      float hprev = Hf[idx];
      float rr = sigf(gi[s][0][v] + bir + gh[s][0][v] + bhr);
      float zz = sigf(gi[s][1][v] + biz + gh[s][1][v] + bhz);
      float nn = tanhf(gi[s][2][v] + bin + rr * (gh[s][2][v] + bhn));
      float hnew = (1.0f - zz) * nn + zz * hprev;
      outf[idx]  = hnew;
      outbf[idx] = f2bf(hnew);
    }
}

// ---------------------------------------------------------------------------
// fused LSTM aggregator + SAGE output
//   block = 256 thr (8 waves) owns 32 node rows; 16 timesteps; c kept in
//   C-layout registers; h kept in LDS (bf16, A-fragment friendly)
//   out = fd@fsW^T + fsb + h16@fnW^T + fnb
// ---------------------------------------------------------------------------
__global__ __launch_bounds__(256) void lstm_sage_kernel(
    const float* __restrict__ fs, const float* __restrict__ fd,
    const int* __restrict__ nbr,
    const unsigned short* __restrict__ Wih, const unsigned short* __restrict__ Whh,
    const float* __restrict__ bih, const float* __restrict__ bhh,
    const unsigned short* __restrict__ fsW, const float* __restrict__ fsb,
    const unsigned short* __restrict__ fnW, const float* __restrict__ fnb,
    float* __restrict__ out) {
  constexpr int LDH = 136;                          // 272B row stride: bank rotate, 16B aligned
  __shared__ __align__(16) unsigned short xld[32 * LDH];
  __shared__ __align__(16) unsigned short hld[32 * LDH];
  const int tid = threadIdx.x, lane = tid & 31, wave = tid >> 5;
  const int row0 = blockIdx.x * 32;

  for (int i = tid; i < 32 * LDH; i += 256) hld[i] = 0;

  const int col = (wave << 4) + (lane & 15);        // hidden index 0..127
  const float bi  = bih[col]          + bhh[col];
  const float bff = bih[kH + col]     + bhh[kH + col];
  const float bg  = bih[2 * kH + col] + bhh[2 * kH + col];
  const float bo  = bih[3 * kH + col] + bhh[3 * kH + col];

  v8f c0 = (v8f)0.0f, c1 = (v8f)0.0f;
  __syncthreads();

  const int gr = tid >> 3;                // gather: row 0..31
  const int gc = (tid & 7) << 4;          // 16-col chunk

  for (int t = 0; t < kD; ++t) {
    {   // gather x_t = fs[nbr[:, t]] -> bf16 LDS
      int j = nbr[(row0 + gr) * kD + t];
      const float* sp = fs + (size_t)j * kH + gc;
      unsigned short* dp = &xld[gr * LDH + gc];
#pragma unroll
      for (int q = 0; q < 16; ++q) dp[q] = f2bf(sp[q]);
    }
    __syncthreads();

    v8f acc[2][4];
#pragma unroll
    for (int s = 0; s < 2; ++s)
#pragma unroll
      for (int g = 0; g < 4; ++g) acc[s][g] = (v8f)0.0f;

#pragma unroll
    for (int kb = 0; kb < 4; ++kb) {
      Frag ax0 = load_a_frag(&xld[0], LDH, kb, lane);
      Frag ax1 = load_a_frag(&xld[16 * LDH], LDH, kb, lane);
      Frag ah0 = load_a_frag(&hld[0], LDH, kb, lane);
      Frag ah1 = load_a_frag(&hld[16 * LDH], LDH, kb, lane);
#pragma unroll
      for (int g = 0; g < 4; ++g) {
        Frag bihf = load_b_frag(Wih + (size_t)(g * kH + (wave << 4)) * kH, kH, kb, lane);
        Frag bhhf = load_b_frag(Whh + (size_t)(g * kH + (wave << 4)) * kH, kH, kb, lane);
        acc[0][g] = wmma_bf16(ax0, bihf, acc[0][g]);
        acc[1][g] = wmma_bf16(ax1, bihf, acc[1][g]);
        acc[0][g] = wmma_bf16(ah0, bhhf, acc[0][g]);
        acc[1][g] = wmma_bf16(ah1, bhhf, acc[1][g]);
      }
    }
    __syncthreads();        // all reads of hld/xld complete

#pragma unroll
    for (int s = 0; s < 2; ++s) {
      v8f& cc = s ? c1 : c0;
#pragma unroll
      for (int v = 0; v < 8; ++v) {
        float iv = sigf(acc[s][0][v] + bi);
        float fv = sigf(acc[s][1][v] + bff);
        float gv = tanhf(acc[s][2][v] + bg);
        float ov = sigf(acc[s][3][v] + bo);
        float cv = fv * cc[v] + iv * gv;
        cc[v] = cv;
        float hv = ov * tanhf(cv);
        int r = v + ((lane >> 4) << 3) + (s << 4);
        hld[r * LDH + col] = f2bf(hv);
      }
    }
    __syncthreads();        // h_t visible before next step
  }

  {   // stage fd rows (bf16) into xld
    const float* sp = fd + (size_t)(row0 + gr) * kH + gc;
    unsigned short* dp = &xld[gr * LDH + gc];
#pragma unroll
    for (int q = 0; q < 16; ++q) dp[q] = f2bf(sp[q]);
  }
  __syncthreads();

  v8f o0 = (v8f)0.0f, o1 = (v8f)0.0f;
#pragma unroll
  for (int kb = 0; kb < 4; ++kb) {
    Frag ad0 = load_a_frag(&xld[0], LDH, kb, lane);
    Frag ad1 = load_a_frag(&xld[16 * LDH], LDH, kb, lane);
    Frag ah0 = load_a_frag(&hld[0], LDH, kb, lane);
    Frag ah1 = load_a_frag(&hld[16 * LDH], LDH, kb, lane);
    Frag bs = load_b_frag(fsW + (size_t)(wave << 4) * kH, kH, kb, lane);
    Frag bn = load_b_frag(fnW + (size_t)(wave << 4) * kH, kH, kb, lane);
    o0 = wmma_bf16(ad0, bs, o0);
    o0 = wmma_bf16(ah0, bn, o0);
    o1 = wmma_bf16(ad1, bs, o1);
    o1 = wmma_bf16(ah1, bn, o1);
  }
  float bb = fsb[col] + fnb[col];
#pragma unroll
  for (int v = 0; v < 8; ++v) {
    int r = v + ((lane >> 4) << 3);
    out[(size_t)(row0 + r) * kH + col]      = o0[v] + bb;
    out[(size_t)(row0 + 16 + r) * kH + col] = o1[v] + bb;
  }
}

// ---------------------------------------------------------------------------
extern "C" void kernel_launch(void* const* d_in, const int* in_sizes, int n_in,
                              void* d_out, int out_size, void* d_ws, size_t ws_size,
                              hipStream_t stream) {
  (void)in_sizes; (void)n_in; (void)out_size; (void)ws_size;
  const int*   act_ids  = (const int*)d_in[0];
  const int*   dur_ids  = (const int*)d_in[1];
  const int*   act2_ids = (const int*)d_in[2];
  const int*   nbr_a2d  = (const int*)d_in[3];
  const int*   nbr_d2a  = (const int*)d_in[4];
  const int*   src2     = (const int*)d_in[5];
  const int*   dst2     = (const int*)d_in[6];
  const int*   gids     = (const int*)d_in[7];
  const float* emb_act  = (const float*)d_in[8];
  const float* emb_dur  = (const float*)d_in[9];
  const float* emb_act2 = (const float*)d_in[10];
  const float* transform_W = (const float*)d_in[11];
  const float* transform_b = (const float*)d_in[12];
  const float* ggc_W  = (const float*)d_in[13];
  const float* ggc_b  = (const float*)d_in[14];
  const float* gru_Wih = (const float*)d_in[15];
  const float* gru_Whh = (const float*)d_in[16];
  const float* gru_bih = (const float*)d_in[17];
  const float* gru_bhh = (const float*)d_in[18];
  const float* hn_g = (const float*)d_in[19];
  const float* hn_b = (const float*)d_in[20];
  const float* lstm_Wih = (const float*)d_in[21];
  const float* lstm_Whh = (const float*)d_in[22];
  const float* lstm_bih = (const float*)d_in[23];
  const float* lstm_bhh = (const float*)d_in[24];
  const float* fcs_W = (const float*)d_in[25];
  const float* fcs_b = (const float*)d_in[26];
  const float* fcn_W = (const float*)d_in[27];
  const float* fcn_b = (const float*)d_in[28];
  const float* n1_g = (const float*)d_in[29];
  const float* n1_b = (const float*)d_in[30];
  const float* n3_g = (const float*)d_in[31];
  const float* n3_b = (const float*)d_in[32];
  const float* mh_W1 = (const float*)d_in[33];
  const float* mh_b1 = (const float*)d_in[34];
  const float* mh_W2 = (const float*)d_in[35];
  const float* mh_b2 = (const float*)d_in[36];
  const float* mo_W1 = (const float*)d_in[37];
  const float* mo_b1 = (const float*)d_in[38];
  const float* mo_W2 = (const float*)d_in[39];
  const float* mo_b2 = (const float*)d_in[40];
  const float* mlp_W1 = (const float*)d_in[41];
  const float* mlp_b1 = (const float*)d_in[42];
  const float* mlp_W2 = (const float*)d_in[43];
  const float* mlp_b2 = (const float*)d_in[44];
  const float* cls_W = (const float*)d_in[45];
  const float* cls_b = (const float*)d_in[46];

  size_t off = 0;
  auto allocf = [&](size_t elems) -> float* {
    float* p = (float*)((char*)d_ws + off);
    off += ((elems * 4 + 255) / 256) * 256;
    return p;
  };
  auto allocu = [&](size_t elems) -> unsigned short* {
    unsigned short* p = (unsigned short*)((char*)d_ws + off);
    off += ((elems * 2 + 255) / 256) * 256;
    return p;
  };

  float* ha   = allocf(kNH);
  float* hd   = allocf(kNH);
  float* na   = allocf(kNH);
  float* nd   = allocf(kNH);
  float* res  = allocf(kNH);
  float* hhA  = allocf(kNH);
  float* hhB  = allocf(kNH);
  float* mbuf = allocf(kNH);
  float* agg  = allocf(kNH);
  float* sums1 = allocf(kB * kH + kB);  float* cnt1 = sums1 + kB * kH;
  float* sums2 = allocf(kB * kH + kB);  float* cnt2 = sums2 + kB * kH;
  float* o1a = allocf(kB * kH);
  float* o1  = allocf(kB * kH);
  float* o2a = allocf(kB * kH);
  float* o2  = allocf(kB * kH);
  float* hc  = allocf(kB * 512);
  float* p1  = allocf(kB * 256);
  float* p2  = allocf(kB * kH);

  unsigned short* concat_bf = allocu((size_t)kN * 256);
  unsigned short* hhbfA = allocu(kNH);
  unsigned short* hhbfB = allocu(kNH);
  unsigned short* abf   = allocu(kNH);
  unsigned short* wT  = allocu(kH * 2 * kH);
  unsigned short* wG  = allocu(kH * kH);
  unsigned short* wGi = allocu(3 * kH * kH);
  unsigned short* wGh = allocu(3 * kH * kH);
  unsigned short* wLi = allocu((size_t)2 * 2 * 4 * kH * kH);
  unsigned short* wLh = allocu((size_t)2 * 2 * 4 * kH * kH);
  unsigned short* wFs = allocu((size_t)2 * 2 * kH * kH);
  unsigned short* wFn = allocu((size_t)2 * 2 * kH * kH);

  auto cvt = [&](const float* s, unsigned short* t, int n) {
    cvt_bf16_kernel<<<(n + 255) / 256, 256, 0, stream>>>(s, t, n);
  };
  cvt(transform_W, wT, kH * 2 * kH);
  cvt(ggc_W, wG, kH * kH);
  cvt(gru_Wih, wGi, 3 * kH * kH);
  cvt(gru_Whh, wGh, 3 * kH * kH);
  cvt(lstm_Wih, wLi, 2 * 2 * 4 * kH * kH);
  cvt(lstm_Whh, wLh, 2 * 2 * 4 * kH * kH);
  cvt(fcs_W, wFs, 2 * 2 * kH * kH);
  cvt(fcn_W, wFn, 2 * 2 * kH * kH);

  // embeddings -> ha(=h_act), hd(=h_dur), concat bf16
  embed_kernel<<<kNH / 256, 256, 0, stream>>>(act_ids, dur_ids, act2_ids, emb_act,
                                              emb_dur, emb_act2, ha, hd, concat_bf);

  // transform: res = h_new (f32) + hhbfA (bf16)
  gemm_bias_kernel<<<kN / 32, 256, 0, stream>>>(concat_bf, wT, transform_b,
                                                res, hhbfA, 2 * kH, kH);

  // GatedGraphConv: 2 steps
  unsigned short* hbf_cur = hhbfA;
  const float* hf_cur = res;
  float* hf_out[2] = {hhA, hhB};
  unsigned short* hbf_out[2] = {hhbfB, hhbfA};
  for (int step = 0; step < 2; ++step) {
    gemm_bias_kernel<<<kN / 32, 256, 0, stream>>>(hbf_cur, wG, ggc_b,
                                                  mbuf, (unsigned short*)nullptr, kH, kH);
    zero_kernel<<<kNH / 256, 256, 0, stream>>>(agg, kNH);
    scatter_add_kernel<<<(unsigned)((size_t)kE2 * kH / 256), 256, 0, stream>>>(mbuf, src2, dst2, agg);
    cvt(agg, abf, kNH);
    gru_kernel<<<kN / 32, 256, 0, stream>>>(abf, hbf_cur, hf_cur, wGi, wGh,
                                            gru_bih, gru_bhh, hf_out[step], hbf_out[step]);
    hbf_cur = hbf_out[step];
    hf_cur  = hf_out[step];
  }
  // h_new = lrelu(LN(hh + res)) -> mbuf ; hg2_mean = seg_mean(h_new) -> sums1
  ln_lrelu_kernel<<<kN, 128, 0, stream>>>(hf_cur, res, hn_g, hn_b, mbuf);
  zero_kernel<<<(kB * kH + kB + 255) / 256, 256, 0, stream>>>(sums1, kB * kH + kB);
  seg_accum_kernel<<<kNH / 256, 256, 0, stream>>>(mbuf, gids, sums1, cnt1);
  seg_div_kernel<<<(kB * kH + 255) / 256, 256, 0, stream>>>(sums1, cnt1);

  // hetero branch: L=2 layers of (SAGE-LSTM a2d, d2a) + norms + residuals
  for (int l = 0; l < 2; ++l) {
    size_t wOff  = (size_t)(l * 2) * 4 * kH * kH;       // per (l,dir) 4H*H
    size_t bOff  = (size_t)(l * 2) * 4 * kH;
    size_t fOff  = (size_t)(l * 2) * kH * kH;
    size_t fbOff = (size_t)(l * 2) * kH;
    // nd: fs=ha, fd=hd, dir 0
    lstm_sage_kernel<<<kN / 32, 256, 0, stream>>>(
        ha, hd, nbr_a2d,
        wLi + wOff, wLh + wOff, lstm_bih + bOff, lstm_bhh + bOff,
        wFs + fOff, fcs_b + fbOff, wFn + fOff, fcn_b + fbOff, nd);
    // na: fs=hd, fd=ha, dir 1
    lstm_sage_kernel<<<kN / 32, 256, 0, stream>>>(
        hd, ha, nbr_d2a,
        wLi + wOff + (size_t)4 * kH * kH, wLh + wOff + (size_t)4 * kH * kH,
        lstm_bih + bOff + 4 * kH, lstm_bhh + bOff + 4 * kH,
        wFs + fOff + (size_t)kH * kH, fcs_b + fbOff + kH,
        wFn + fOff + (size_t)kH * kH, fcn_b + fbOff + kH, na);
    const float* n1g0 = n1_g + (size_t)(l * 2 + 0) * kH;
    const float* n1b0 = n1_b + (size_t)(l * 2 + 0) * kH;
    const float* n1g1 = n1_g + (size_t)(l * 2 + 1) * kH;
    const float* n1b1 = n1_b + (size_t)(l * 2 + 1) * kH;
    const float* n3g0 = n3_g + (size_t)(l * 2 + 0) * kH;
    const float* n3b0 = n3_b + (size_t)(l * 2 + 0) * kH;
    const float* n3g1 = n3_g + (size_t)(l * 2 + 1) * kH;
    const float* n3b1 = n3_b + (size_t)(l * 2 + 1) * kH;
    ln_lrelu_kernel<<<kN, 128, 0, stream>>>(na, (const float*)nullptr, n1g0, n1b0, na);
    ln_lrelu_kernel<<<kN, 128, 0, stream>>>(nd, (const float*)nullptr, n1g1, n1b1, nd);
    ln_lrelu_kernel<<<kN, 128, 0, stream>>>(ha, na, n3g0, n3b0, ha);
    ln_lrelu_kernel<<<kN, 128, 0, stream>>>(hd, nd, n3g1, n3b1, hd);
  }
  // hg_mean = seg_mean(ha) + seg_mean(hd) -> sums2
  zero_kernel<<<(kB * kH + kB + 255) / 256, 256, 0, stream>>>(sums2, kB * kH + kB);
  seg_accum_kernel<<<kNH / 256, 256, 0, stream>>>(ha, gids, sums2, cnt2);
  seg_accum_kernel<<<kNH / 256, 256, 0, stream>>>(hd, gids, sums2, (float*)nullptr);
  seg_div_kernel<<<(kB * kH + 255) / 256, 256, 0, stream>>>(sums2, cnt2);

  // heads (exact fp32)
  auto lin = [&](const float* X, const float* W, const float* b, float* out,
                 int M, int K, int O, int relu) {
    linear_kernel<<<(M * O + 255) / 256, 256, 0, stream>>>(X, W, b, out, M, K, O, relu);
  };
  lin(sums1, mo_W1, mo_b1, o1a, kB, kH, kH, 1);
  lin(o1a,   mo_W2, mo_b2, o1,  kB, kH, kH, 1);
  lin(sums2, mh_W1, mh_b1, o2a, kB, kH, kH, 1);
  lin(o2a,   mh_W2, mh_b2, o2,  kB, kH, kH, 1);
  concat4_kernel<<<(kB * 512 + 255) / 256, 256, 0, stream>>>(o1, sums1, o2, sums2, hc);
  lin(hc, mlp_W1, mlp_b1, p1, kB, 512, 256, 1);
  lin(p1, mlp_W2, mlp_b2, p2, kB, 256, kH, 1);
  lin(p2, cls_W, cls_b, (float*)d_out, kB, kH, kNC, 0);
}
